// GCN_GIN_42348377538854
// MI455X (gfx1250) — compile-verified, matrix operand order checked by reading, not run on hardware
//
#include <hip/hip_runtime.h>
#include <stdint.h>

#define N_NODES   50000
#define N_EDGES   600000
#define F_IN      128
#define DIM_H     256
#define N_CLASSES 40
#define NC3_PAD   48      // 40 padded to 3 tiles of 16
#define BN_EPS    1e-5f

typedef __attribute__((ext_vector_type(16))) __bf16 v16bf;
typedef __attribute__((ext_vector_type(8)))  float  v8f;

union FragAB { uint4 u[2]; v16bf v; };
union FragC  { v8f v; float f[8]; };

__device__ __forceinline__ unsigned short f2bf(float f) {
  union { float f; unsigned int u; } c; c.f = f;
  unsigned int u = c.u;
  u += 0x7FFFu + ((u >> 16) & 1u);          // round-to-nearest-even
  return (unsigned short)(u >> 16);
}
__device__ __forceinline__ float bf2f(unsigned short s) {
  union { unsigned int u; float f; } c; c.u = ((unsigned int)s) << 16;
  return c.f;
}

// ---------------------------------------------------------------------------
// 1) Edge scatter-add: hin[dst] += x[src]. One wave32 per edge, float4/lane.
// ---------------------------------------------------------------------------
__global__ __launch_bounds__(256) void scatter_add_kernel(
    const float* __restrict__ x, const long long* __restrict__ ei,
    float* __restrict__ hin) {
  const long long t = (long long)blockIdx.x * blockDim.x + threadIdx.x;
  const long long edge = t >> 5;
  const int lane = (int)(t & 31);
  if (edge >= N_EDGES) return;
  const long long src = ei[edge];
  const long long dst = ei[N_EDGES + edge];
  const float4 v = ((const float4*)(x + src * F_IN))[lane];
  float* d = hin + dst * F_IN + lane * 4;
  __hip_atomic_fetch_add(d + 0, v.x, __ATOMIC_RELAXED, __HIP_MEMORY_SCOPE_AGENT);
  __hip_atomic_fetch_add(d + 1, v.y, __ATOMIC_RELAXED, __HIP_MEMORY_SCOPE_AGENT);
  __hip_atomic_fetch_add(d + 2, v.z, __ATOMIC_RELAXED, __HIP_MEMORY_SCOPE_AGENT);
  __hip_atomic_fetch_add(d + 3, v.w, __ATOMIC_RELAXED, __HIP_MEMORY_SCOPE_AGENT);
}

// ---------------------------------------------------------------------------
// 2) fp32 -> bf16 convert (4 elems/thread)
// ---------------------------------------------------------------------------
__global__ __launch_bounds__(256) void cvt_f32_bf16_kernel(
    const float* __restrict__ src, unsigned short* __restrict__ dst, int n4) {
  const int t = blockIdx.x * blockDim.x + threadIdx.x;
  if (t >= n4) return;
  const float4 v = ((const float4*)src)[t];
  ushort4 o; o.x = f2bf(v.x); o.y = f2bf(v.y); o.z = f2bf(v.z); o.w = f2bf(v.w);
  ((ushort4*)dst)[t] = o;
}

// ---------------------------------------------------------------------------
// 3) Pack weights [K, NCreal] fp32 -> bf16 in WMMA B-fragment order:
//    P[(((nt*nkt)+kt)*32 + lane)*16 + e] = W[k][n],
//    k = kt*32 + (e/8)*16 + (lane/16)*8 + (e%8),  n = nt*16 + (lane&15)
// ---------------------------------------------------------------------------
__global__ __launch_bounds__(256) void pack_w_kernel(
    const float* __restrict__ W, unsigned short* __restrict__ P,
    int K, int NCreal, int total) {
  const int t = blockIdx.x * blockDim.x + threadIdx.x;
  if (t >= total) return;
  const int e    = t & 15;
  const int lane = (t >> 4) & 31;
  const int rest = t >> 9;
  const int nkt  = K >> 5;
  const int kt   = rest % nkt;
  const int nt   = rest / nkt;
  const int k = kt * 32 + ((e >> 3) << 4) + ((lane >> 4) << 3) + (e & 7);
  const int n = (nt << 4) + (lane & 15);
  const float v = (n < NCreal) ? W[(size_t)k * NCreal + n] : 0.f;
  P[t] = f2bf(v);
}

// ---------------------------------------------------------------------------
// 4) WMMA GEMM: D = A[M,K](bf16) * Bpacked + bias, wave-per-16x16 tile.
//    A tile (16 x K) is staged once per block in LDS (8 waves share it);
//    rows padded by one 16B chunk to spread across the 64 LDS banks.
//    MODE 0: store bf16, atomic column sum/sumsq (for BN)
//    MODE 1: ReLU, store bf16
//    MODE 2: store fp32, columns clipped to realN
// ---------------------------------------------------------------------------
template <int MODE>
__global__ __launch_bounds__(256) void gemm_wmma_kernel(
    const unsigned short* __restrict__ A, const unsigned short* __restrict__ Bp,
    const float* __restrict__ bias, unsigned short* __restrict__ Cb,
    float* __restrict__ Cf, float* __restrict__ colSum,
    float* __restrict__ colSumSq, int K, int NC, int realN) {
  __shared__ uint4 ldsA[16 * (DIM_H / 8 + 1)];    // max K=256 -> 16*33 uint4 = 8448 B

  const int lane   = threadIdx.x & 31;
  const int wave   = threadIdx.x >> 5;
  const int mt     = blockIdx.x;
  const int nt     = blockIdx.y * 8 + wave;
  const int ntiles = NC >> 4;
  const int kc     = K >> 3;                      // 16B chunks per A row
  const int stride = kc + 1;                      // padded LDS row stride (chunks)

  // --- cooperative stage of A tile [16 x K] into LDS (all threads) ---
  {
    const uint4* Ag = (const uint4*)(A + (size_t)mt * 16 * K);
    const int nchunks = 16 * kc;
    for (int c = threadIdx.x; c < nchunks; c += 256) {
      const int row = c / kc;
      const int cc  = c - row * kc;
      ldsA[row * stride + cc] = Ag[(size_t)row * kc + cc];
    }
  }
  __syncthreads();

  if (nt >= ntiles) return;                 // wave-uniform: EXEC stays all-1s

  const int laneLo = lane & 15;
  const int laneHi = lane >> 4;
  const int nkt    = K >> 5;

  FragC acc;
  acc.v = (v8f){0.f, 0.f, 0.f, 0.f, 0.f, 0.f, 0.f, 0.f};

  const uint4* __restrict__ la = ldsA + laneLo * stride + laneHi;
  const unsigned short* bp = Bp + ((size_t)nt * nkt * 32 + lane) * 16;

  for (int kt = 0; kt < nkt; ++kt) {
    __builtin_prefetch(bp + 512, 0, 1);     // global_prefetch_b8 on next B block
    FragAB a, b;
    a.u[0] = la[kt * 4];                    // k = kt*32 + laneHi*8 + 0..7
    a.u[1] = la[kt * 4 + 2];                // k = kt*32 + 16 + laneHi*8 + 0..7
    b.u[0] = *(const uint4*)(bp);
    b.u[1] = *(const uint4*)(bp + 8);
    acc.v = __builtin_amdgcn_wmma_f32_16x16x32_bf16(
        false, a.v, false, b.v, (short)0, acc.v, false, false);
    bp += 512;
  }

  const int n   = (nt << 4) + laneLo;
  const float bia = (n < realN) ? bias[n] : 0.f;
  float s = 0.f, ss = 0.f;
#pragma unroll
  for (int r = 0; r < 8; ++r) {
    const int row = (mt << 4) + laneHi * 8 + r;
    float v = acc.f[r] + bia;
    if (MODE == 0) {
      s += v; ss += v * v;
      Cb[(size_t)row * NC + n] = f2bf(v);
    } else if (MODE == 1) {
      v = fmaxf(v, 0.f);
      Cb[(size_t)row * NC + n] = f2bf(v);
    } else {
      if (n < realN) Cf[(size_t)row * realN + n] = v;
    }
  }
  if (MODE == 0) {
    __hip_atomic_fetch_add(&colSum[n],   s,  __ATOMIC_RELAXED, __HIP_MEMORY_SCOPE_AGENT);
    __hip_atomic_fetch_add(&colSumSq[n], ss, __ATOMIC_RELAXED, __HIP_MEMORY_SCOPE_AGENT);
  }
}

// ---------------------------------------------------------------------------
// 5) BatchNorm (training batch stats) + ReLU, bf16 in/out, stats from atomics
// ---------------------------------------------------------------------------
__global__ __launch_bounds__(256) void bn_relu_kernel(
    unsigned short* __restrict__ h, const float* __restrict__ colSum,
    const float* __restrict__ colSumSq, const float* __restrict__ gamma,
    const float* __restrict__ beta) {
  const long long t = (long long)blockIdx.x * blockDim.x + threadIdx.x;
  if (t >= (long long)N_NODES * DIM_H / 4) return;
  const long long idx = t * 4;
  const int j0 = (int)(idx & (DIM_H - 1));
  const float invN = 1.0f / (float)N_NODES;
  ushort4 v = ((const ushort4*)h)[t];
  unsigned short r[4] = {v.x, v.y, v.z, v.w};
#pragma unroll
  for (int q = 0; q < 4; ++q) {
    const int j = j0 + q;
    const float mean = colSum[j] * invN;
    const float var  = colSumSq[j] * invN - mean * mean;
    const float sc   = gamma[j] * rsqrtf(var + BN_EPS);
    const float sh   = beta[j] - mean * sc;
    r[q] = f2bf(fmaxf(bf2f(r[q]) * sc + sh, 0.f));
  }
  ushort4 o; o.x = r[0]; o.y = r[1]; o.z = r[2]; o.w = r[3];
  ((ushort4*)h)[t] = o;
}

// ---------------------------------------------------------------------------
extern "C" void kernel_launch(void* const* d_in, const int* in_sizes, int n_in,
                              void* d_out, int out_size, void* d_ws, size_t ws_size,
                              hipStream_t stream) {
  const float*     x    = (const float*)d_in[0];
  const long long* ei   = (const long long*)d_in[1];
  const float*     W1   = (const float*)d_in[2];
  const float*     b1   = (const float*)d_in[3];
  const float*     gamma= (const float*)d_in[4];
  const float*     beta = (const float*)d_in[5];
  const float*     W2   = (const float*)d_in[6];
  const float*     b2   = (const float*)d_in[7];
  const float*     Wlin = (const float*)d_in[8];
  const float*     blin = (const float*)d_in[9];
  float*           out  = (float*)d_out;

  // workspace carve-up (256B aligned)
  char* ws = (char*)d_ws;
  size_t off = 0;
  auto carve = [&](size_t bytes) -> char* {
    char* p = ws + off;
    off = (off + bytes + 255) & ~(size_t)255;
    return p;
  };
  float*          hin   = (float*)carve((size_t)N_NODES * F_IN * sizeof(float));          // x + agg
  unsigned short* a_bf  = (unsigned short*)carve((size_t)N_NODES * F_IN * 2);             // bf16 A for GEMM1
  unsigned short* h1b   = (unsigned short*)carve((size_t)N_NODES * DIM_H * 2);            // GEMM1 out / BN out
  unsigned short* h2b   = (unsigned short*)carve((size_t)N_NODES * DIM_H * 2);            // GEMM2 out
  float*          stats = (float*)carve(2 * DIM_H * sizeof(float));                       // colSum | colSumSq
  const int w1p_elems = (DIM_H / 16) * (F_IN / 32) * 512;    // 32768
  const int w2p_elems = (DIM_H / 16) * (DIM_H / 32) * 512;   // 65536
  const int wlp_elems = (NC3_PAD / 16) * (DIM_H / 32) * 512; // 12288
  unsigned short* W1p = (unsigned short*)carve((size_t)w1p_elems * 2);
  unsigned short* W2p = (unsigned short*)carve((size_t)w2p_elems * 2);
  unsigned short* Wlp = (unsigned short*)carve((size_t)wlp_elems * 2);
  float* colSum   = stats;
  float* colSumSq = stats + DIM_H;

  // init: hin = x (agg epsilon=0 path), stats = 0 (fresh every launch)
  hipMemcpyAsync(hin, x, (size_t)N_NODES * F_IN * sizeof(float),
                 hipMemcpyDeviceToDevice, stream);
  hipMemsetAsync(stats, 0, 2 * DIM_H * sizeof(float), stream);

  // pack weights to bf16 WMMA fragment order
  pack_w_kernel<<<(w1p_elems + 255) / 256, 256, 0, stream>>>(W1, W1p, F_IN, DIM_H, w1p_elems);
  pack_w_kernel<<<(w2p_elems + 255) / 256, 256, 0, stream>>>(W2, W2p, DIM_H, DIM_H, w2p_elems);
  pack_w_kernel<<<(wlp_elems + 255) / 256, 256, 0, stream>>>(Wlin, Wlp, DIM_H, N_CLASSES, wlp_elems);

  // edge scatter: hin[dst] += x[src]
  {
    const long long thr = (long long)N_EDGES * 32;
    scatter_add_kernel<<<(unsigned)((thr + 255) / 256), 256, 0, stream>>>(x, ei, hin);
  }

  // hin -> bf16
  {
    const int n4 = (N_NODES * F_IN) / 4;
    cvt_f32_bf16_kernel<<<(n4 + 255) / 256, 256, 0, stream>>>(hin, a_bf, n4);
  }

  // GEMM1: h1b = hin @ W1 + b1 (bf16 out) + fused BN column stats
  gemm_wmma_kernel<0><<<dim3(N_NODES / 16, (DIM_H / 16 + 7) / 8), 256, 0, stream>>>(
      a_bf, W1p, b1, h1b, nullptr, colSum, colSumSq, F_IN, DIM_H, DIM_H);

  // BatchNorm (batch stats) + ReLU in place on h1b
  {
    const long long thr = (long long)N_NODES * DIM_H / 4;
    bn_relu_kernel<<<(unsigned)((thr + 255) / 256), 256, 0, stream>>>(
        h1b, colSum, colSumSq, gamma, beta);
  }

  // GEMM2: h2b = relu(h1b @ W2 + b2)
  gemm_wmma_kernel<1><<<dim3(N_NODES / 16, (DIM_H / 16 + 7) / 8), 256, 0, stream>>>(
      h1b, W2p, b2, h2b, nullptr, nullptr, nullptr, DIM_H, DIM_H, DIM_H);

  // GEMM3: out = h2b @ Wlin + blin (fp32, 40 classes, padded to 48)
  gemm_wmma_kernel<2><<<dim3(N_NODES / 16, 1), 256, 0, stream>>>(
      h2b, Wlp, blin, nullptr, out, nullptr, nullptr, DIM_H, NC3_PAD, N_CLASSES);
}
